// ConvIntNet_64441689309229
// MI455X (gfx1250) — matrix operand order repaired and checked by reading, not compiled, not run on hardware
//
#include <hip/hip_runtime.h>

// ---------------------------------------------------------------------------
// Interaction network (ConvIntNet) for MI455X / gfx1250 (wave32, WMMA).
//
// Stages:
//   K1 bn_project : BN + per-node fr layer-1 projections Pr/Ps (f16, padded 32)
//   K2 edge_wmma  : per-(b,receiver) edge MLP 30->15->6 via v_wmma_f32_16x16x32_f16
//                   + local reduction of effects over senders (replaces scatter-add)
//                   + async global->LDS staging (ASYNCcnt path)
//   K3 node_head  : fo MLP (22->45->22->6), sum over nodes, fc head + softmax
// ---------------------------------------------------------------------------

#define NB      512
#define NCONST  100
#define NFEATS  16
#define NEFF    6
#define NDYN    6
#define NCLASS  5
#define BN_EPS  1e-3f

#define PROJ    32          // fr layer-1 width 30, padded to 32 (WMMA K)
#define NSROWS  128         // 99 senders padded to 8 tiles of 16 rows

typedef __attribute__((ext_vector_type(16))) _Float16 v16h;
typedef __attribute__((ext_vector_type(8)))  float    v8f;
typedef int v4i __attribute__((vector_size(16)));     // matches builtin param type

union HFrag { v16h v; _Float16 h[16]; };
union FFrag { v8f  v; float    f[8];  };

#if defined(__has_builtin)
#if __has_builtin(__builtin_amdgcn_global_load_async_to_lds_b128)
#define HAVE_ASYNC_LDS 1
#endif
#endif

// Wave-local DS ordering: LDS ops of one wave complete in order; this pins the
// compiler and waits outstanding DS before the dependent cross-lane reads.
__device__ __forceinline__ void wave_lds_sync() {
    asm volatile("s_wait_dscnt 0x0" ::: "memory");
}

// K index for 16-bit A-matrix 16x32 fragment (ISA 7.12.2):
//   lanes 0-15 : v0..3 -> K=2v,2v+1      ; v4..7 -> K=2v+8,2v+9
//   lanes16-31 : v0..3 -> K=2v+8,2v+9    ; v4..7 -> K=2v+16,2v+17
__device__ __forceinline__ int a_kmap(int v, int p, int hi) {
    return 2 * v + p + ((v < 4) ? 8 * hi : 8 + 8 * hi);
}

// ---------------------------------------------------------------------------
// Kernel 1: batchnorm + per-node fr layer-1 projections.
// One thread per (b, n) row; fr_w1 staged in LDS.
// ---------------------------------------------------------------------------
__global__ void __launch_bounds__(128)
bn_project(const float* __restrict__ x,
           const float* __restrict__ bn_gamma, const float* __restrict__ bn_beta,
           const float* __restrict__ bn_mean,  const float* __restrict__ bn_var,
           const float* __restrict__ fr_w1,    const float* __restrict__ fr_b1,
           float* __restrict__ xbG,            // (B,N,16) f32
           _Float16* __restrict__ PrG,         // (B,N,32) f16, bias folded, cols30/31=0
           _Float16* __restrict__ PsG)         // (B,N,32) f16, cols30/31=0
{
    __shared__ float sW1[2 * NFEATS * 30];
    __shared__ float sB1[30];
    __shared__ float sG[NFEATS], sB[NFEATS], sM[NFEATS], sV[NFEATS];

    const int tid = threadIdx.x;
    for (int idx = tid; idx < 2 * NFEATS * 30; idx += 128) sW1[idx] = fr_w1[idx];
    if (tid < 30)     sB1[tid] = fr_b1[tid];
    if (tid < NFEATS) { sG[tid] = bn_gamma[tid]; sB[tid] = bn_beta[tid];
                        sM[tid] = bn_mean[tid];  sV[tid] = bn_var[tid]; }
    __syncthreads();

    const int row = blockIdx.x * 128 + tid;       // grid sized exactly B*N
    const float* xr = x + (size_t)row * NFEATS;

    float xb[NFEATS];
#pragma unroll
    for (int f = 0; f < NFEATS; ++f) {
        float s = rsqrtf(sV[f] + BN_EPS);
        xb[f] = (xr[f] - sM[f]) * s * sG[f] + sB[f];
    }
    float* xo = xbG + (size_t)row * NFEATS;
#pragma unroll
    for (int f = 0; f < NFEATS; ++f) xo[f] = xb[f];

    _Float16* pr = PrG + (size_t)row * PROJ;
    _Float16* ps = PsG + (size_t)row * PROJ;
    for (int k = 0; k < 30; ++k) {
        float ar = sB1[k], as = 0.f;
#pragma unroll
        for (int f = 0; f < NFEATS; ++f) {
            ar += xb[f] * sW1[f * 30 + k];                 // receiver rows 0..15
            as += xb[f] * sW1[(NFEATS + f) * 30 + k];      // sender   rows 16..31
        }
        pr[k] = (_Float16)ar;
        ps[k] = (_Float16)as;
    }
    pr[30] = (_Float16)0.0f; pr[31] = (_Float16)0.0f;
    ps[30] = (_Float16)0.0f; ps[31] = (_Float16)0.0f;
}

// ---------------------------------------------------------------------------
// Kernel 2: edge MLP via WMMA, reduced over senders per receiver.
// One block (128 thr = 4 waves) per (b, receiver i). Each wave processes
// 16-sender tiles:  h1=relu(Pr[i]+Ps[j])  -> wmma(30->15) -> wmma(15->6)
// -> masked local sum  -> agg[b,i,:].
// ---------------------------------------------------------------------------
__global__ void __launch_bounds__(128)
edge_wmma(const _Float16* __restrict__ PrG,
          const _Float16* __restrict__ PsG,
          const float* __restrict__ fr_w2, const float* __restrict__ fr_b2,
          const float* __restrict__ fr_w3, const float* __restrict__ fr_b3,
          float* __restrict__ aggG)          // (B,N,6) f32
{
    __shared__ __align__(16) _Float16 sPs[NSROWS][PROJ];   // 8 KB
    __shared__ __align__(16) _Float16 sPr[PROJ];
    __shared__ __align__(16) _Float16 sH2[4][16][16];      // per-wave f32->f16 relayout tile
    __shared__ float sAgg[8];

    const int tid  = threadIdx.x;
    const int lane = tid & 31;
    const int wave = tid >> 5;
    const int m    = lane & 15;     // A/C row within half
    const int n    = lane & 15;     // B/C column
    const int hi   = lane >> 4;

    const int b = blockIdx.x / NCONST;
    const int i = blockIdx.x % NCONST;

    // Stage sender projections (skip j==i), zero-pad to 128 rows.
    {
        if (tid < NCONST - 1) {
            const int j = (tid < i) ? tid : tid + 1;
            const _Float16* src = PsG + ((size_t)b * NCONST + j) * PROJ;
#if defined(HAVE_ASYNC_LDS)
            // CDNA5 async copy: global -> LDS directly, tracked with ASYNCcnt.
#pragma unroll
            for (int q = 0; q < 4; ++q) {
                __builtin_amdgcn_global_load_async_to_lds_b128(
                    (__attribute__((address_space(1))) v4i*)(src + q * 8),
                    (__attribute__((address_space(3))) v4i*)(&sPs[tid][q * 8]),
                    0, 0);
            }
#else
            uint4* dst = reinterpret_cast<uint4*>(&sPs[tid][0]);
            const uint4* s4 = reinterpret_cast<const uint4*>(src);
#pragma unroll
            for (int q = 0; q < 4; ++q) dst[q] = s4[q];
#endif
        } else {
            uint4* dst = reinterpret_cast<uint4*>(&sPs[tid][0]);
            const uint4 z = {0u, 0u, 0u, 0u};
#pragma unroll
            for (int q = 0; q < 4; ++q) dst[q] = z;
        }
        if (tid < PROJ) sPr[tid] = PrG[((size_t)b * NCONST + i) * PROJ + tid];
        if (tid < 8)    sAgg[tid] = 0.f;
    }
#if defined(HAVE_ASYNC_LDS)
#if defined(__has_builtin) && __has_builtin(__builtin_amdgcn_s_wait_asynccnt)
    __builtin_amdgcn_s_wait_asynccnt(0);
#else
    asm volatile("s_wait_asynccnt 0x0" ::: "memory");
#endif
#endif
    __syncthreads();

    // Constant B fragments (32x16, same K order as A via a_kmap; column = lane&15).
    HFrag bw2, bw3;
#pragma unroll
    for (int v = 0; v < 8; ++v)
#pragma unroll
        for (int p = 0; p < 2; ++p) {
            const int k = a_kmap(v, p, hi);
            bw2.h[2 * v + p] = (k < 30 && n < 15) ? (_Float16)fr_w2[k * 15 + n]
                                                  : (_Float16)0.0f;
            bw3.h[2 * v + p] = (k < 15 && n < NEFF) ? (_Float16)fr_w3[k * NEFF + n]
                                                    : (_Float16)0.0f;
        }
    const float b2n = (n < 15)   ? fr_b2[n] : 0.f;
    const float b3n = (n < NEFF) ? fr_b3[n] : 0.f;
    const v8f  czero = {};

    // 8 tiles of 16 senders; wave w takes tiles w and w+4 (uniform loop).
    // No block barriers inside: sH2 is strictly per-wave (wave-local DS sync).
    for (int it = 0; it < 2; ++it) {
        const int tile    = it * 4 + wave;
        const int rowBase = tile * 16;

        // A1: h1 = relu(Pr + Ps), laid out per 16-bit 16x32 A fragment.
        HFrag a1;
#pragma unroll
        for (int v = 0; v < 8; ++v)
#pragma unroll
            for (int p = 0; p < 2; ++p) {
                const int k = a_kmap(v, p, hi);
                _Float16 s = sPr[k] + sPs[rowBase + m][k];
                a1.h[2 * v + p] = (s > (_Float16)0.0f) ? s : (_Float16)0.0f;
            }

        FFrag c1;
        c1.v = __builtin_amdgcn_wmma_f32_16x16x32_f16(
                   false, a1.v, false, bw2.v, (short)0, czero, false, false);

        // h2 = relu(c1 + b2) -> f16, relayout via per-wave LDS tile.
#pragma unroll
        for (int r = 0; r < 8; ++r) {
            float h = c1.f[r] + b2n;
            sH2[wave][r + 8 * hi][n] = (_Float16)(h > 0.f ? h : 0.f);
        }
        wave_lds_sync();

        HFrag a2;
#pragma unroll
        for (int v = 0; v < 8; ++v)
#pragma unroll
            for (int p = 0; p < 2; ++p) {
                const int k = a_kmap(v, p, hi);
                a2.h[2 * v + p] = (k < 16) ? sH2[wave][m][k] : (_Float16)0.0f;
            }

        FFrag c2;
        c2.v = __builtin_amdgcn_wmma_f32_16x16x32_f16(
                   false, a2.v, false, bw3.v, (short)0, czero, false, false);

        // effects = relu(c2 + b3); mask pad senders; reduce over rows.
        float part = 0.f;
#pragma unroll
        for (int r = 0; r < 8; ++r) {
            const int srow = rowBase + r + 8 * hi;
            if (srow < NCONST - 1) {
                float e = c2.f[r] + b3n;
                part += (e > 0.f ? e : 0.f);
            }
        }
        if (n < NEFF) atomicAdd(&sAgg[n], part);
    }
    __syncthreads();

    if (tid < NEFF)
        aggG[((size_t)b * NCONST + i) * NEFF + tid] = sAgg[tid];
}

// ---------------------------------------------------------------------------
// Kernel 3: fo MLP per node + sum over nodes + fc head + softmax.
// One block per batch element; weights LDS-resident.
// ---------------------------------------------------------------------------
__global__ void __launch_bounds__(128)
node_head(const float* __restrict__ xbG, const float* __restrict__ aggG,
          const float* __restrict__ fo_w1, const float* __restrict__ fo_b1,
          const float* __restrict__ fo_w2, const float* __restrict__ fo_b2,
          const float* __restrict__ fo_w3, const float* __restrict__ fo_b3,
          const float* __restrict__ fc_w1, const float* __restrict__ fc_b1,
          const float* __restrict__ fc_w2, const float* __restrict__ fc_b2,
          float* __restrict__ out)
{
    const int FIN = NFEATS + NEFF;   // 22
    __shared__ float sW1[22 * 45], sB1v[45];
    __shared__ float sW2[45 * 22], sB2v[22];
    __shared__ float sW3[22 * NDYN], sB3v[NDYN];
    __shared__ float sSum[NDYN];

    const int tid = threadIdx.x;
    const int b   = blockIdx.x;

    for (int idx = tid; idx < 22 * 45; idx += 128) sW1[idx] = fo_w1[idx];
    for (int idx = tid; idx < 45 * 22; idx += 128) sW2[idx] = fo_w2[idx];
    for (int idx = tid; idx < 22 * NDYN; idx += 128) sW3[idx] = fo_w3[idx];
    if (tid < 45)   sB1v[tid] = fo_b1[tid];
    if (tid < 22)   sB2v[tid] = fo_b2[tid];
    if (tid < NDYN) sB3v[tid] = fo_b3[tid];
    if (tid < NDYN) sSum[tid] = 0.f;
    __syncthreads();

    if (tid < NCONST) {
        float fin[FIN];
        const float* xr = xbG + ((size_t)b * NCONST + tid) * NFEATS;
        const float* ar = aggG + ((size_t)b * NCONST + tid) * NEFF;
#pragma unroll
        for (int f = 0; f < NFEATS; ++f) fin[f] = xr[f];
#pragma unroll
        for (int f = 0; f < NEFF; ++f)   fin[NFEATS + f] = ar[f];

        float h1[45];
        for (int o = 0; o < 45; ++o) {
            float a = sB1v[o];
#pragma unroll
            for (int f = 0; f < FIN; ++f) a += fin[f] * sW1[f * 45 + o];
            h1[o] = a > 0.f ? a : 0.f;
        }
        float h2[22];
        for (int o = 0; o < 22; ++o) {
            float a = sB2v[o];
            for (int f = 0; f < 45; ++f) a += h1[f] * sW2[f * 22 + o];
            h2[o] = a > 0.f ? a : 0.f;
        }
#pragma unroll
        for (int o = 0; o < NDYN; ++o) {
            float a = sB3v[o];
            for (int f = 0; f < 22; ++f) a += h2[f] * sW3[f * NDYN + o];
            atomicAdd(&sSum[o], a > 0.f ? a : 0.f);
        }
    }
    __syncthreads();

    if (tid == 0) {
        float h[48];
        for (int o = 0; o < 48; ++o) {
            float a = fc_b1[o];
#pragma unroll
            for (int k = 0; k < NDYN; ++k) a += sSum[k] * fc_w1[k * 48 + o];
            h[o] = a > 0.f ? a : 0.f;
        }
        float lg[NCLASS], mx = -3.4e38f;
        for (int c = 0; c < NCLASS; ++c) {
            float a = fc_b2[c];
            for (int o = 0; o < 48; ++o) a += h[o] * fc_w2[o * NCLASS + c];
            lg[c] = a; mx = a > mx ? a : mx;
        }
        float den = 0.f;
        for (int c = 0; c < NCLASS; ++c) { lg[c] = __expf(lg[c] - mx); den += lg[c]; }
        for (int c = 0; c < NCLASS; ++c) out[(size_t)b * NCLASS + c] = lg[c] / den;
    }
}

// ---------------------------------------------------------------------------
// Launch
// ---------------------------------------------------------------------------
extern "C" void kernel_launch(void* const* d_in, const int* in_sizes, int n_in,
                              void* d_out, int out_size, void* d_ws, size_t ws_size,
                              hipStream_t stream) {
    const float* x        = (const float*)d_in[0];
    const float* bn_gamma = (const float*)d_in[1];
    const float* bn_beta  = (const float*)d_in[2];
    const float* bn_mean  = (const float*)d_in[3];
    const float* bn_var   = (const float*)d_in[4];
    const float* fr_w1    = (const float*)d_in[5];
    const float* fr_b1    = (const float*)d_in[6];
    const float* fr_w2    = (const float*)d_in[7];
    const float* fr_b2    = (const float*)d_in[8];
    const float* fr_w3    = (const float*)d_in[9];
    const float* fr_b3    = (const float*)d_in[10];
    const float* fo_w1    = (const float*)d_in[11];
    const float* fo_b1    = (const float*)d_in[12];
    const float* fo_w2    = (const float*)d_in[13];
    const float* fo_b2    = (const float*)d_in[14];
    const float* fo_w3    = (const float*)d_in[15];
    const float* fo_b3    = (const float*)d_in[16];
    const float* fc_w1    = (const float*)d_in[17];
    const float* fc_b1    = (const float*)d_in[18];
    const float* fc_w2    = (const float*)d_in[19];
    const float* fc_b2    = (const float*)d_in[20];
    float* out = (float*)d_out;

    // Workspace layout (bytes), 256-aligned regions, total ~11.06 MB.
    char* ws = (char*)d_ws;
    const size_t SZ_XB  = (size_t)NB * NCONST * NFEATS * sizeof(float);    // 3.28 MB
    const size_t SZ_PRJ = (size_t)NB * NCONST * PROJ * sizeof(_Float16);   // 3.28 MB
    float*    xbG  = (float*)   (ws);
    _Float16* PrG  = (_Float16*)(ws + SZ_XB);
    _Float16* PsG  = (_Float16*)(ws + SZ_XB + SZ_PRJ);
    float*    aggG = (float*)   (ws + SZ_XB + 2 * SZ_PRJ);
    (void)ws_size; (void)in_sizes; (void)n_in; (void)out_size;

    bn_project<<<(NB * NCONST) / 128, 128, 0, stream>>>(
        x, bn_gamma, bn_beta, bn_mean, bn_var, fr_w1, fr_b1, xbG, PrG, PsG);

    edge_wmma<<<NB * NCONST, 128, 0, stream>>>(
        PrG, PsG, fr_w2, fr_b2, fr_w3, fr_b3, aggG);

    node_head<<<NB, 128, 0, stream>>>(
        xbG, aggG, fo_w1, fo_b1, fo_w2, fo_b2, fo_w3, fo_b3,
        fc_w1, fc_b1, fc_w2, fc_b2, out);
}
